// Attention_68959994905247
// MI455X (gfx1250) — matrix operand order repaired and checked by reading, not compiled, not run on hardware
//
#include <hip/hip_runtime.h>

// ---------------------------------------------------------------------------
// Local-window attention (7x11) for B=8, H=32, W=64 (N=2048), C=512, heads=8,
// head_dim=64 on MI455X (gfx1250, wave32, bf16 WMMA 16x16x32, f32 accum).
//
// K0a : pack weights  f32 -> bf16, B-operand lane order (2x b128 per tile)
// K0b : pack x        f32 -> bf16, A-operand lane order (2x b128 per tile)
// K1  : qkv GEMM (pure bf16 loop) -> q (scaled), k, vT (transposed)
// K2  : banded flash attention; K/V tiles double-buffered into LDS via the
//       Tensor Data Mover (tensor_load_to_lds, in-order TENSORcnt ping-pong)
// K3  : output projection + bias (f32 out)
//
// Workspace: q/k/vT/o (4 x 16 MB) + packed weights (2 MB) + packed x (16 MB)
//          = 82 MB.
// ---------------------------------------------------------------------------

#define NTOK 2048
#define CDIM 512
#define NHEAD 8
#define HD 64
#define IMW 64
#define IMH 32
#define NEG_INF (-1e30f)

typedef __attribute__((ext_vector_type(16))) __bf16 v16bf;
typedef __attribute__((ext_vector_type(8)))  float  v8f;
typedef __attribute__((ext_vector_type(4)))  unsigned int v4u;
typedef __attribute__((ext_vector_type(8)))  int v8i;
typedef __attribute__((ext_vector_type(4)))  int v4i;

#if defined(__AMDGCN__) && __has_builtin(__builtin_amdgcn_tensor_load_to_lds)
#define HAVE_TDM 1
#endif

__device__ __forceinline__ v8f wmma_bf16(v16bf a, v16bf b, v8f c) {
  return __builtin_amdgcn_wmma_f32_16x16x32_bf16(
      false, a, false, b, (short)0, c, false, false);
}

// A-operand (16x32 bf16), ISA 7.12.2: lanes 0-15: M=lane, K={k0..+7,k0+16..+23};
// lanes 16-31: +8 on both K groups.
__device__ __forceinline__ v16bf load_a_bf16(const __bf16* __restrict__ src,
                                             int ld, int k0, int lane) {
  const __bf16* p = src + (size_t)(lane & 15) * ld + k0 + ((lane & 16) ? 8 : 0);
  v16bf a;
#pragma unroll
  for (int i = 0; i < 8; ++i) {
    a[i]     = p[i];
    a[i + 8] = p[i + 16];
  }
  return a;
}

// B-operand (32x16) with B[k][n] = src[n][k] (src row-major, k contiguous):
// lanes 0-15: column n=lane, K={k0..+7,k0+16..+23}; lanes 16-31: +8 on both K groups.
__device__ __forceinline__ v16bf load_bt_bf16(const __bf16* __restrict__ src,
                                              int ld, int k0, int lane) {
  const __bf16* p = src + (size_t)(lane & 15) * ld + k0 + ((lane & 16) ? 8 : 0);
  v16bf b;
#pragma unroll
  for (int i = 0; i < 8; ++i) {
    b[i]     = p[i];
    b[i + 8] = p[i + 16];
  }
  return b;
}

#ifdef HAVE_TDM
// Tensor Data Mover: 2D tile (tile_d0 contiguous elems x tile_d1 rows of
// bf16, row stride stride0 elems) from global -> LDS. D# per ISA ch.8.
// Uniform per wave (descriptor in SGPRs); EXEC ignored by tensor ops.
__device__ __forceinline__ void tdm_load_2d(const void* gptr, unsigned lds_off,
                                            unsigned tile_d0, unsigned tile_d1,
                                            unsigned long long stride0) {
  unsigned long long ga = (unsigned long long)(uintptr_t)gptr;
  v4u g0;
  g0[0] = 1u;                                             // count=1 (user D#)
  g0[1] = lds_off;                                        // lds_addr [63:32]
  g0[2] = (unsigned)ga;                                   // global_addr lo
  g0[3] = (unsigned)((ga >> 32) & 0x1FFFFFFu) | (2u << 30);  // addr hi | type=2
  const unsigned td0 = tile_d0, td1 = tile_d1;            // tensor dims = tile
  v8i g1;
  g1[0] = (int)(1u << 16);                                // data_size=1 (2B)
  g1[1] = (int)((td0 & 0xFFFFu) << 16);                   // tensor_dim0 lo
  g1[2] = (int)((td0 >> 16) | ((td1 & 0xFFFFu) << 16));   // dim0 hi | dim1 lo
  g1[3] = (int)((td1 >> 16) | (tile_d0 << 16));           // dim1 hi | tile0
  g1[4] = (int)tile_d1;                                   // tile1 | tile2=0
  g1[5] = (int)(unsigned)(stride0 & 0xFFFFFFFFu);         // dim0_stride lo
  g1[6] = (int)(unsigned)((stride0 >> 32) & 0xFFFFu);     // stride hi | s1 lo
  g1[7] = 0;
  v4i z4 = {0, 0, 0, 0};
#if defined(__clang_major__) && __clang_major__ >= 23
  v8i z8 = {0, 0, 0, 0, 0, 0, 0, 0};
  __builtin_amdgcn_tensor_load_to_lds(g0, g1, z4, z4, z8, 0);
#else
  __builtin_amdgcn_tensor_load_to_lds(g0, g1, z4, z4, 0);
#endif
}
#endif  // HAVE_TDM

// ---------------------------------------------------------------------------
// K0a: pack weights into bf16 B-operand lane order, tile = nt*(CDIM/32)+kt.
// ---------------------------------------------------------------------------
__global__ __launch_bounds__(256) void pack_w_k(const float* __restrict__ w,
                                                __bf16* __restrict__ wpk) {
  const int lane = threadIdx.x & 31;
  const int wv   = threadIdx.x >> 5;
  const int tile = blockIdx.x * 8 + wv;
  const int kt   = tile & 15;   // CDIM/32 = 16
  const int nt   = tile >> 4;
  const float* p = w + (size_t)(nt * 16 + (lane & 15)) * CDIM + kt * 32 +
                   ((lane & 16) ? 8 : 0);
  v16bf b;
#pragma unroll
  for (int i = 0; i < 8; ++i) {
    b[i]     = (__bf16)p[i];
    b[i + 8] = (__bf16)p[i + 16];
  }
  *reinterpret_cast<v16bf*>(wpk + ((size_t)tile * 32 + lane) * 16) = b;
}

// ---------------------------------------------------------------------------
// K0b: pack x into bf16 A-operand lane order, tile = mt*(CDIM/32)+kt.
// x rows are reused 24x by K1; converting/swizzling once removes all cvt
// VALU from the GEMM hot loop and halves A-fetch bytes.
// ---------------------------------------------------------------------------
__global__ __launch_bounds__(256) void pack_x_k(const float* __restrict__ x,
                                                __bf16* __restrict__ xpk) {
  const int lane = threadIdx.x & 31;
  const int wv   = threadIdx.x >> 5;
  const int tile = blockIdx.x * 8 + wv;
  const int kt   = tile & 15;
  const int mt   = tile >> 4;
  const float* p = x + (size_t)(mt * 16 + (lane & 15)) * CDIM + kt * 32 +
                   ((lane & 16) ? 8 : 0);
  __builtin_prefetch(p + CDIM, 0, 0);
  v16bf a;
#pragma unroll
  for (int i = 0; i < 8; ++i) {
    a[i]     = (__bf16)p[i];
    a[i + 8] = (__bf16)p[i + 16];
  }
  *reinterpret_cast<v16bf*>(xpk + ((size_t)tile * 32 + lane) * 16) = a;
}

// ---------------------------------------------------------------------------
// K1: qkv = x @ qkv_w.T, both operands pre-packed bf16; emit q(scaled), k, vT.
// Wave = 16(M) x 64(N); inner loop = 2xb128(A) + 4*(2xb128(B) + wmma).
// ---------------------------------------------------------------------------
__global__ __launch_bounds__(256) void qkv_gemm_k(
    const __bf16* __restrict__ xpk, const __bf16* __restrict__ wpk,
    __bf16* __restrict__ q, __bf16* __restrict__ kk, __bf16* __restrict__ vT) {
  const int lane = threadIdx.x & 31;
  const int wv   = threadIdx.x >> 5;
  const int mt   = blockIdx.y * 8 + wv;
  const int gm0  = mt * 16;
  const int nc0  = blockIdx.x * 64;
  const int t    = nc0 >> 9;            // 0=q 1=k 2=v (uniform)
  const int hh   = (nc0 >> 6) & 7;
  const int b    = gm0 >> 11;
  const int tok0 = gm0 & (NTOK - 1);
  const int nt0  = nc0 >> 4;
  const size_t bh = (size_t)b * NHEAD + hh;

  v8f acc[4] = {};

#pragma unroll 1
  for (int kt = 0; kt < 16; ++kt) {
    const __bf16* ap = xpk + (((size_t)mt * 16 + kt) * 32 + lane) * 16;
    v16bf a = *reinterpret_cast<const v16bf*>(ap);
#pragma unroll
    for (int cc = 0; cc < 4; ++cc) {
      const __bf16* bp = wpk + (((size_t)(nt0 + cc) * 16 + kt) * 32 + lane) * 16;
      v16bf bt = *reinterpret_cast<const v16bf*>(bp);
      acc[cc] = wmma_bf16(a, bt, acc[cc]);
    }
  }

  const int ncol = lane & 15;
  const int rb   = (lane & 16) ? 8 : 0;
#pragma unroll
  for (int cc = 0; cc < 4; ++cc) {
    const int dloc = cc * 16 + ncol;
    if (t == 2) {
      alignas(16) unsigned short pk[8];
#pragma unroll
      for (int r = 0; r < 8; ++r)
        pk[r] = __builtin_bit_cast(unsigned short, (__bf16)acc[cc][r]);
      size_t base = (bh * HD + dloc) * (size_t)NTOK + tok0 + rb;
      *reinterpret_cast<int4*>(vT + base) = *reinterpret_cast<const int4*>(pk);
    } else {
      __bf16* dst = (t == 0) ? q : kk;
      const float sc = (t == 0) ? 0.125f : 1.0f;  // head_dim^-0.5
#pragma unroll
      for (int r = 0; r < 8; ++r) {
        size_t addr = (bh * NTOK + tok0 + rb + r) * (size_t)HD + dloc;
        dst[addr] = (__bf16)(acc[cc][r] * sc);
      }
    }
  }
}

// ---------------------------------------------------------------------------
// K2: banded flash attention, one wave per (b, head, 16-query tile).
// Key band = image rows hq-3..hq+3; |dw|<=5 mask in VALU; fully-masked
// 32-key steps skipped (uniform). With TDM: double-buffered K/V staging —
// issue next step's pair, s_wait_tensorcnt(2) (in-order) to consume current.
// ---------------------------------------------------------------------------
__global__ __launch_bounds__(256) void attn_k(
    const __bf16* __restrict__ q, const __bf16* __restrict__ kk,
    const __bf16* __restrict__ vT, __bf16* __restrict__ o) {
  __shared__ alignas(16) unsigned short plds[8][16 * 32];  // P re-layout stage
#ifdef HAVE_TDM
  __shared__ alignas(16) __bf16 ktile_s[8][2][32 * HD];    // 2 x 4 KB / wave
  __shared__ alignas(16) __bf16 vtile_s[8][2][HD * 32];    // 2 x 4 KB / wave
#endif

  const int lane = threadIdx.x & 31;
  const int wv   = threadIdx.x >> 5;
  const int gw   = blockIdx.x * 8 + wv;
  const int qt   = gw & 127;
  const int hh   = (gw >> 7) & 7;
  const int b    = gw >> 10;
  const int n0   = qt * 16;
  const int hq   = n0 >> 6;
  const int wq0  = n0 & 63;
  const size_t bh = (size_t)b * NHEAD + hh;

  const int ncol = lane & 15;
  const int rb   = (lane & 16) ? 8 : 0;

  const __bf16* qbase = q + (bh * NTOK + n0) * HD;
  v16bf aq0 = load_a_bf16(qbase, HD, 0, lane);
  v16bf aq1 = load_a_bf16(qbase, HD, 32, lane);

  v8f oacc[4] = {};
  float mrow[8], lrow[8];
#pragma unroll
  for (int r = 0; r < 8; ++r) { mrow[r] = NEG_INF; lrow[r] = 0.0f; }

  const int kn_lo = ((hq >= 3) ? (hq - 3) : 0) * IMW;
  const int kn_hi = (((hq + 4) <= IMH) ? (hq + 4) : IMH) * IMW;

  // Next non-fully-masked 32-key step after knprev (uniform scalar scan).
  auto next_valid = [&](int knprev) -> int {
    for (int kx = knprev + 32; kx < kn_hi; kx += 32) {
      const int off = kx & 63;
      if (!(off + 31 < wq0 - 5 || off > wq0 + 20)) return kx;
    }
    return -1;
  };

#ifdef HAVE_TDM
  unsigned klds[2], vlds[2];
#pragma unroll
  for (int s = 0; s < 2; ++s) {
    klds[s] = (unsigned)(uintptr_t)&ktile_s[wv][s][0];
    vlds[s] = (unsigned)(uintptr_t)&vtile_s[wv][s][0];
  }
  auto stage = [&](int knx, int buf) {
    tdm_load_2d(kk + (bh * NTOK + knx) * HD, klds[buf], HD, 32, HD);
    tdm_load_2d(vT + bh * HD * (size_t)NTOK + knx, vlds[buf], 32, HD, NTOK);
  };
#endif

  int kn  = next_valid(kn_lo - 32);
  int cur = 0;
#ifdef HAVE_TDM
  if (kn >= 0) stage(kn, 0);
#endif

  while (kn >= 0) {
    const int knn = next_valid(kn);

    v8f s0, s1;
#ifdef HAVE_TDM
    if (knn >= 0) {
      stage(knn, cur ^ 1);                     // prefetch next pair
      __builtin_amdgcn_s_wait_tensorcnt(2);    // in-order: current pair done
    } else {
      __builtin_amdgcn_s_wait_tensorcnt(0);
    }
    const __bf16* kbl = &ktile_s[wv][cur][0];
    s0 = wmma_bf16(aq0, load_bt_bf16(kbl, HD, 0, lane), v8f{});
    s0 = wmma_bf16(aq1, load_bt_bf16(kbl, HD, 32, lane), s0);
    const __bf16* kbl1 = kbl + 16 * HD;
    s1 = wmma_bf16(aq0, load_bt_bf16(kbl1, HD, 0, lane), v8f{});
    s1 = wmma_bf16(aq1, load_bt_bf16(kbl1, HD, 32, lane), s1);
#else
    const __bf16* kb = kk + (bh * NTOK + kn) * HD;
    s0 = wmma_bf16(aq0, load_bt_bf16(kb, HD, 0, lane), v8f{});
    s0 = wmma_bf16(aq1, load_bt_bf16(kb, HD, 32, lane), s0);
    const __bf16* kb1 = kb + 16 * HD;
    s1 = wmma_bf16(aq0, load_bt_bf16(kb1, HD, 0, lane), v8f{});
    s1 = wmma_bf16(aq1, load_bt_bf16(kb1, HD, 32, lane), s1);
#endif

    // ---- mask + online softmax ----
    const int wk0 = (kn + ncol) & 63;
    const int wk1 = (kn + 16 + ncol) & 63;
    bool a0m[8], a1m[8];
    float mx[8];
#pragma unroll
    for (int r = 0; r < 8; ++r) {
      const int wqr = wq0 + rb + r;
      a0m[r] = (wk0 - wqr <= 5) && (wqr - wk0 <= 5);
      a1m[r] = (wk1 - wqr <= 5) && (wqr - wk1 <= 5);
      const float v0 = a0m[r] ? s0[r] : NEG_INF;
      const float v1 = a1m[r] ? s1[r] : NEG_INF;
      mx[r] = fmaxf(v0, v1);
    }
#pragma unroll
    for (int d = 1; d <= 8; d <<= 1) {
#pragma unroll
      for (int r = 0; r < 8; ++r) mx[r] = fmaxf(mx[r], __shfl_xor(mx[r], d));
    }
    float p0[8], p1[8], rs[8];
#pragma unroll
    for (int r = 0; r < 8; ++r) {
      const float nm    = fmaxf(mrow[r], mx[r]);
      const float alpha = __expf(mrow[r] - nm);
      mrow[r] = nm;
      p0[r] = a0m[r] ? __expf(s0[r] - nm) : 0.0f;
      p1[r] = a1m[r] ? __expf(s1[r] - nm) : 0.0f;
      rs[r] = p0[r] + p1[r];
      lrow[r] *= alpha;
#pragma unroll
      for (int c = 0; c < 4; ++c) oacc[c][r] *= alpha;
    }
#pragma unroll
    for (int d = 1; d <= 8; d <<= 1) {
#pragma unroll
      for (int r = 0; r < 8; ++r) rs[r] += __shfl_xor(rs[r], d);
    }
#pragma unroll
    for (int r = 0; r < 8; ++r) lrow[r] += rs[r];

    // ---- P: C-layout -> A-layout through per-wave LDS (DS in-order/wave) ----
    unsigned short* mp = plds[wv];
#pragma unroll
    for (int r = 0; r < 8; ++r) {
      const int m = rb + r;
      mp[m * 32 + ncol]      = __builtin_bit_cast(unsigned short, (__bf16)p0[r]);
      mp[m * 32 + 16 + ncol] = __builtin_bit_cast(unsigned short, (__bf16)p1[r]);
    }
    v16bf pa = load_a_bf16(reinterpret_cast<const __bf16*>(mp), 32, 0, lane);

    // ---- O(16x64) += P(16x32) @ V(32x64) ----
#pragma unroll
    for (int c = 0; c < 4; ++c) {
#ifdef HAVE_TDM
      const __bf16* vb = &vtile_s[wv][cur][(c * 16) * 32];
      oacc[c] = wmma_bf16(pa, load_bt_bf16(vb, 32, 0, lane), oacc[c]);
#else
      const __bf16* vb = vT + (bh * HD + c * 16) * (size_t)NTOK + kn;
      oacc[c] = wmma_bf16(pa, load_bt_bf16(vb, NTOK, 0, lane), oacc[c]);
#endif
    }

    kn = knn;
    cur ^= 1;
  }

  // ---- normalize, store O in (b, n, C) layout ----
#pragma unroll
  for (int c = 0; c < 4; ++c) {
#pragma unroll
    for (int r = 0; r < 8; ++r) {
      const float val = oacc[c][r] / lrow[r];
      size_t addr = ((size_t)b * NTOK + n0 + rb + r) * CDIM +
                    hh * HD + c * 16 + ncol;
      o[addr] = (__bf16)val;
    }
  }
}

// ---------------------------------------------------------------------------
// K3: out = O @ proj_w.T + bias (f32), packed bf16 weights.
// ---------------------------------------------------------------------------
__global__ __launch_bounds__(256) void proj_k(
    const __bf16* __restrict__ a, const __bf16* __restrict__ wpk,
    const float* __restrict__ bias, float* __restrict__ out) {
  const int lane = threadIdx.x & 31;
  const int wv   = threadIdx.x >> 5;
  const int gm0  = (blockIdx.y * 8 + wv) * 16;
  const int nc0  = blockIdx.x * 64;
  const int nt0  = nc0 >> 4;

  v8f acc[4] = {};
  const __bf16* arow = a + (size_t)gm0 * CDIM;

#pragma unroll 1
  for (int kt = 0; kt < 16; ++kt) {
    v16bf av = load_a_bf16(arow, CDIM, kt * 32, lane);
#pragma unroll
    for (int cc = 0; cc < 4; ++cc) {
      const __bf16* bp = wpk + (((size_t)(nt0 + cc) * 16 + kt) * 32 + lane) * 16;
      v16bf bt = *reinterpret_cast<const v16bf*>(bp);
      acc[cc] = wmma_bf16(av, bt, acc[cc]);
    }
  }

  const int ncol = lane & 15;
  const int rb   = (lane & 16) ? 8 : 0;
#pragma unroll
  for (int cc = 0; cc < 4; ++cc) {
    const int col = nc0 + cc * 16 + ncol;
    const float bv = bias[col];
#pragma unroll
    for (int r = 0; r < 8; ++r)
      out[(size_t)(gm0 + rb + r) * CDIM + col] = acc[cc][r] + bv;
  }
}

// ---------------------------------------------------------------------------
extern "C" void kernel_launch(void* const* d_in, const int* in_sizes, int n_in,
                              void* d_out, int out_size, void* d_ws, size_t ws_size,
                              hipStream_t stream) {
  const float* x      = (const float*)d_in[0];  // (8, 2048, 512)
  const float* qkv_w  = (const float*)d_in[1];  // (1536, 512)
  const float* proj_w = (const float*)d_in[2];  // (512, 512)
  const float* proj_b = (const float*)d_in[3];  // (512,)
  float* out = (float*)d_out;                   // (8, 2048, 512)

  const size_t HELEMS = (size_t)8 * NHEAD * NTOK * HD;  // 8,388,608
  __bf16* q_ws  = (__bf16*)d_ws;
  __bf16* k_ws  = q_ws + HELEMS;
  __bf16* vT_ws = k_ws + HELEMS;
  __bf16* o_ws  = vT_ws + HELEMS;                 // (8,2048,512) bf16
  __bf16* wq_pk = o_ws + HELEMS;                  // 1536x512 bf16 packed
  __bf16* wp_pk = wq_pk + (size_t)1536 * CDIM;    // 512x512 bf16 packed
  __bf16* x_pk  = wp_pk + (size_t)CDIM * CDIM;    // 16384x512 bf16 packed
  // total workspace = 82 MB

  pack_w_k<<<dim3(192), 256, 0, stream>>>(qkv_w, wq_pk);    // 96*16 tiles
  pack_w_k<<<dim3(64), 256, 0, stream>>>(proj_w, wp_pk);    // 32*16 tiles
  pack_x_k<<<dim3(2048), 256, 0, stream>>>(x, x_pk);        // 1024*16 tiles
  qkv_gemm_k<<<dim3(24, 128), 256, 0, stream>>>(x_pk, wq_pk, q_ws, k_ws, vT_ws);
  attn_k<<<dim3(1024), 256, 0, stream>>>(q_ws, k_ws, vT_ws, o_ws);
  proj_k<<<dim3(8, 128), 256, 0, stream>>>(o_ws, wp_pk, proj_b, out);
}
